// LinearLoop_21251498180732
// MI455X (gfx1250) — compile-verified, matrix operand order checked by reading, not run on hardware
//
#include <hip/hip_runtime.h>

// ---------------------------------------------------------------------------
// out[B, OUT] = x[B, IN] * W[OUT, IN]^T + b[OUT]    (fp32, 3xBF16-split WMMA)
// B=8192, IN=1024, OUT=1024
// Double-buffered LDS, one barrier per K-step, bf16 hi/lo split for ~fp32
// accuracy at bf16-WMMA throughput.
// ---------------------------------------------------------------------------

#define B_DIM  8192
#define IN_F   1024
#define OUT_F  1024

#define TILE_M 128
#define TILE_N 128
#define TILE_K 32
#define LDS_STRIDE 40   // bf16 elems/row: 32 data + 8 pad => 80B rows (16B aligned)

typedef __attribute__((ext_vector_type(16))) __bf16          v16bf;
typedef __attribute__((ext_vector_type(8)))  float           v8f;
typedef __attribute__((ext_vector_type(4)))  float           v4f;
typedef __attribute__((ext_vector_type(4)))  unsigned short  u16x4;
typedef __attribute__((ext_vector_type(8)))  unsigned short  u16x8;
typedef __attribute__((ext_vector_type(16))) unsigned short  u16x16;

// Native bf16 converts (compiler picks the hw v_cvt path on gfx1250).
__device__ __forceinline__ unsigned short f32_to_bf16(float f) {
  __bf16 h = (__bf16)f;
  return __builtin_bit_cast(unsigned short, h);
}
__device__ __forceinline__ float bf16_to_f32(unsigned short u) {
  return (float)__builtin_bit_cast(__bf16, u);
}

// Issue the 4 float4 global loads for this thread's share of a 128x32 tile.
__device__ __forceinline__ void load_tile_regs(const float* __restrict__ g,
                                               v4f r[4], int tid) {
#pragma unroll
  for (int i = 0; i < 4; ++i) {
    int f   = tid + i * 256;   // float4 id 0..1023
    int row = f >> 3;          // 0..127
    int c4  = f & 7;
    r[i] = *(const v4f*)(g + (size_t)row * IN_F + c4 * 4);
  }
}

// Split registers into bf16 hi/lo planes and store to LDS.
__device__ __forceinline__ void convert_store(const v4f r[4],
                                              unsigned short* __restrict__ hiP,
                                              unsigned short* __restrict__ loP,
                                              int tid) {
#pragma unroll
  for (int i = 0; i < 4; ++i) {
    int f   = tid + i * 256;
    int row = f >> 3;
    int c4  = f & 7;
    u16x4 hi, lo;
#pragma unroll
    for (int j = 0; j < 4; ++j) {
      float v = r[i][j];
      unsigned short h = f32_to_bf16(v);
      hi[j] = h;
      lo[j] = f32_to_bf16(v - bf16_to_f32(h));
    }
    *(u16x4*)(hiP + row * LDS_STRIDE + c4 * 4) = hi;
    *(u16x4*)(loP + row * LDS_STRIDE + c4 * 4) = lo;
  }
}

// 16x32 bf16 fragment load, CDNA5 16-bit A/B VGPR layout:
// lanes 0-15: K 0..7 (v0..v3), K 16..23 (v4..v7) of row base+lane
// lanes 16-31: K 8..15, K 24..31 of row base+lane-16
__device__ __forceinline__ v16bf load_frag(const unsigned short* __restrict__ plane,
                                           int rowBase, int lane) {
  int r  = rowBase + (lane & 15);
  int k0 = (lane >> 4) << 3;          // 0 or 8
  const unsigned short* q = plane + r * LDS_STRIDE + k0;
  u16x8 a = *(const u16x8*)(q);       // K = k0 .. k0+7
  u16x8 b = *(const u16x8*)(q + 16);  // K = k0+16 .. k0+23
  u16x16 c = __builtin_shufflevector(a, b,
      0, 1, 2, 3, 4, 5, 6, 7, 8, 9, 10, 11, 12, 13, 14, 15);
  return __builtin_bit_cast(v16bf, c);
}

__global__ __launch_bounds__(256)
void linear_wmma_bf16x3_kernel(const float* __restrict__ x,
                               const float* __restrict__ W,
                               const float* __restrict__ bias,
                               float* __restrict__ out) {
  // [buf][hi=0/lo=1] ping-pong planes
  __shared__ unsigned short sA[2][2][TILE_M * LDS_STRIDE];
  __shared__ unsigned short sB[2][2][TILE_N * LDS_STRIDE];

  const int tid  = threadIdx.x;
  const int lane = tid & 31;
  const int wave = tid >> 5;

  const int blockN = blockIdx.x * TILE_N;   // output-feature tile
  const int blockM = blockIdx.y * TILE_M;   // batch tile

  // 8 waves: 4 along M (32 rows) x 2 along N (64 cols)
  const int mBase = (wave >> 1) * 32;
  const int nBase = (wave & 1) * 64;

  v8f acc[2][4];
#pragma unroll
  for (int mi = 0; mi < 2; ++mi)
#pragma unroll
    for (int ni = 0; ni < 4; ++ni)
      acc[mi][ni] = (v8f){0.f, 0.f, 0.f, 0.f, 0.f, 0.f, 0.f, 0.f};

  const float* xTile = x + (size_t)blockM * IN_F;
  const float* wTile = W + (size_t)blockN * IN_F;

  // ---- prologue: stage K-tile 0 into buffer 0 ----
  v4f rx[4], rw[4];
  load_tile_regs(xTile, rx, tid);
  load_tile_regs(wTile, rw, tid);
  convert_store(rx, sA[0][0], sA[0][1], tid);
  convert_store(rw, sB[0][0], sB[0][1], tid);
  __syncthreads();

  int buf = 0;
  for (int k = 0; k < IN_F; k += TILE_K) {
    const int kn = k + TILE_K;
    const bool hasNext = kn < IN_F;

    // 1) issue global loads for the next K tile (latency hidden under WMMA)
    if (hasNext) {
      load_tile_regs(xTile + kn, rx, tid);
      load_tile_regs(wTile + kn, rw, tid);
      // prefetch K tile k+2 toward L2 (global_prefetch_b8)
      if (kn + TILE_K < IN_F) {
#pragma unroll
        for (int i = 0; i < 4; ++i) {
          int r = (tid >> 3) + i * 32;
          __builtin_prefetch(xTile + (size_t)r * IN_F + kn + TILE_K, 0, 1);
          __builtin_prefetch(wTile + (size_t)r * IN_F + kn + TILE_K, 0, 1);
        }
      }
    }

    // 2) fragment loads from current buffer
    v16bf aHi[2], aLo[2], bHi[4], bLo[4];
#pragma unroll
    for (int mi = 0; mi < 2; ++mi) {
      aHi[mi] = load_frag(sA[buf][0], mBase + mi * 16, lane);
      aLo[mi] = load_frag(sA[buf][1], mBase + mi * 16, lane);
    }
#pragma unroll
    for (int ni = 0; ni < 4; ++ni) {
      bHi[ni] = load_frag(sB[buf][0], nBase + ni * 16, lane);
      bLo[ni] = load_frag(sB[buf][1], nBase + ni * 16, lane);
    }

    // 3) 24 WMMAs: hi*hi + hi*lo + lo*hi  (fp32 accumulate)
#pragma unroll
    for (int mi = 0; mi < 2; ++mi) {
#pragma unroll
      for (int ni = 0; ni < 4; ++ni) {
        acc[mi][ni] = __builtin_amdgcn_wmma_f32_16x16x32_bf16(
            false, aHi[mi], false, bHi[ni], (short)0, acc[mi][ni], false, false);
        acc[mi][ni] = __builtin_amdgcn_wmma_f32_16x16x32_bf16(
            false, aHi[mi], false, bLo[ni], (short)0, acc[mi][ni], false, false);
        acc[mi][ni] = __builtin_amdgcn_wmma_f32_16x16x32_bf16(
            false, aLo[mi], false, bHi[ni], (short)0, acc[mi][ni], false, false);
      }
    }

    // 4) convert + store next tile into the other buffer (overlaps WMMA)
    if (hasNext) {
      convert_store(rx, sA[buf ^ 1][0], sA[buf ^ 1][1], tid);
      convert_store(rw, sB[buf ^ 1][0], sB[buf ^ 1][1], tid);
    }

    // 5) single barrier per iteration covers both hazards:
    //    - this iter's reads of buf finish before next iter writes buf
    //    - this iter's writes of buf^1 happen after prev iter's reads (prev barrier)
    __syncthreads();
    buf ^= 1;
  }

  // Epilogue: C/D layout — lane%16 = column, VGPR i = row i (+8 for lanes>=16)
#pragma unroll
  for (int mi = 0; mi < 2; ++mi) {
#pragma unroll
    for (int ni = 0; ni < 4; ++ni) {
      int row0 = blockM + mBase + mi * 16 + ((lane >> 4) << 3);
      int col  = blockN + nBase + ni * 16 + (lane & 15);
      float bv = bias[col];
#pragma unroll
      for (int i = 0; i < 8; ++i) {
        out[(size_t)(row0 + i) * OUT_F + col] = acc[mi][ni][i] + bv;
      }
    }
  }
}

extern "C" void kernel_launch(void* const* d_in, const int* in_sizes, int n_in,
                              void* d_out, int out_size, void* d_ws, size_t ws_size,
                              hipStream_t stream) {
  (void)in_sizes; (void)n_in; (void)out_size; (void)d_ws; (void)ws_size;
  const float* x    = (const float*)d_in[0];
  const float* W    = (const float*)d_in[1];
  const float* bias = (const float*)d_in[2];
  float* out        = (float*)d_out;

  dim3 grid(OUT_F / TILE_N, B_DIM / TILE_M);  // (8, 64)
  dim3 block(256);
  linear_wmma_bf16x3_kernel<<<grid, block, 0, stream>>>(x, W, bias, out);
}
